// ContentOnlyModel_81750407512130
// MI455X (gfx1250) — matrix operand order repaired and checked by reading, not compiled, not run on hardware
//
#include <hip/hip_runtime.h>

typedef __attribute__((ext_vector_type(16))) _Float16 v16h;
typedef __attribute__((ext_vector_type(8)))  float    v8f;

#define DT_   768
#define DI_   512
#define H_    128
#define HM_   64
#define NPOS_ 25600          // B*T = 128*200
#define NTOK_ 51200          // pos + neg
#define TILES_ (NTOK_/16)    // 3200
#define WAVES_ 2
#define ASTRIDE_ (DT_ + 8)   // f16 elems; row stride = 1552 B (16B aligned)
#define FSTRIDE_ (H_ + 8)    // f16 elems; row stride = 272 B (16B aligned)
#define EPSL 1e-5f

// ---------------------------------------------------------------------------
// fragment builder: two 16-byte chunks -> v16h (8 VGPRs)
// ---------------------------------------------------------------------------
union Frag16 { uint4 u[2]; v16h v; };

__device__ __forceinline__ v16h frag_ab(const _Float16* p0, const _Float16* p1) {
  Frag16 f;
  f.u[0] = *(const uint4*)p0;
  f.u[1] = *(const uint4*)p1;
  return f.v;
}

// prefetch one contiguous byte range at cacheline stride (global_prefetch_b8)
__device__ __forceinline__ void prefetch_range(const void* p, int bytes) {
  const char* c = (const char*)p;
#pragma unroll
  for (int off = 0; off < bytes; off += 128)
    __builtin_prefetch(c + off, 0, 3);
}

// ---------------------------------------------------------------------------
// gather one table row, LayerNorm it (fp32 stats), emit f16 into LDS A-tile.
// 2 lanes per row: lane covers columns [half*D/2, (half+1)*D/2).
// ---------------------------------------------------------------------------
template<int D>
__device__ __forceinline__ void gather_ln(const float* __restrict__ table, int id,
                                          const float* __restrict__ g,
                                          const float* __restrict__ b,
                                          _Float16* dstrow, int half)
{
  constexpr int C = D / 2;
  const float4* __restrict__ src = (const float4*)(table + (size_t)id * D + half * C);
  float s = 0.f, sq = 0.f;
  for (int i = 0; i < C / 4; ++i) {
    float4 v = src[i];
    s  += v.x + v.y + v.z + v.w;
    sq += v.x*v.x + v.y*v.y + v.z*v.z + v.w*v.w;
  }
  s  += __shfl_xor(s,  1, 32);
  sq += __shfl_xor(sq, 1, 32);
  const float mu  = s * (1.f / D);
  const float var = sq * (1.f / D) - mu * mu;
  const float rs  = rsqrtf(var + EPSL);

  const float4* __restrict__ g4 = (const float4*)(g + half * C);
  const float4* __restrict__ b4 = (const float4*)(b + half * C);
  _Float16* dst = dstrow + half * C;
  for (int i = 0; i < C / 4; ++i) {
    float4 v = src[i], gg = g4[i], bb = b4[i];
    union { _Float16 h[4]; uint2 u; } pk;
    pk.h[0] = (_Float16)((v.x - mu) * rs * gg.x + bb.x);
    pk.h[1] = (_Float16)((v.y - mu) * rs * gg.y + bb.y);
    pk.h[2] = (_Float16)((v.z - mu) * rs * gg.z + bb.z);
    pk.h[3] = (_Float16)((v.w - mu) * rs * gg.w + bb.w);
    *(uint2*)(dst + 4 * i) = pk.u;   // 8-byte DS store
  }
}

// ---------------------------------------------------------------------------
// D(16xNCH*16) += A(16xK, LDS tile) * W^T  via v_wmma_f32_16x16x32_f16.
// A layout per ISA 7.12.2 (16-bit 16x32): lane row = lane&15,
//   K chunks {akb..akb+7} and {akb+16..akb+23}, akb = (lane&16)?8:0.
// B: weights W[out][k] row-major f16 -> lane col n = lane&15 reads the
//   same K chunks as A: {akb..akb+7} and {akb+16..akb+23}.
// ---------------------------------------------------------------------------
template<int K, int NCH, int AST>
__device__ __forceinline__ void wmma_mm(const _Float16* A,
                                        const _Float16* __restrict__ W,
                                        v8f* acc, int lane)
{
  const int n   = lane & 15;
  const int akb = (lane & 16) ? 8 : 0;
  const int bkb = (lane & 16) ? 16 : 0;
  const _Float16* arow = A + n * AST;   // A row index == lane&15
  for (int k0 = 0; k0 < K; k0 += 32) {
    v16h a = frag_ab(arow + k0 + akb, arow + k0 + akb + 16);
#pragma unroll
    for (int nc = 0; nc < NCH; ++nc) {
      const _Float16* bp = W + (size_t)(nc * 16 + n) * K + k0 + akb;
      v16h bf = frag_ab(bp, bp + 16);
      acc[nc] = __builtin_amdgcn_wmma_f32_16x16x32_f16(
          false, a, false, bf, (short)0, acc[nc], false, false);
    }
  }
}

// bias + ReLU + f16 pack into fused LDS tile (C/D layout: lane col n, rows rb..rb+7)
template<int NCH, int FST>
__device__ __forceinline__ void store_h(const v8f* acc, const float* __restrict__ bias,
                                        _Float16 (*F)[FST], int coloff, int lane)
{
  const int n  = lane & 15;
  const int rb = (lane & 16) ? 8 : 0;
#pragma unroll
  for (int nc = 0; nc < NCH; ++nc) {
    const float bv = bias[nc * 16 + n];
#pragma unroll
    for (int i = 0; i < 8; ++i) {
      float h = acc[nc][i] + bv;
      F[rb + i][coloff + nc * 16 + n] = (_Float16)(h > 0.f ? h : 0.f);
    }
  }
}

// ---------------------------------------------------------------------------
// weight f32 -> f16 conversion (weights then live in L2 for the main kernel)
// ---------------------------------------------------------------------------
__global__ void cvt_f32_f16(const float* __restrict__ s, _Float16* __restrict__ d, int n) {
  int i = blockIdx.x * blockDim.x + threadIdx.x;
  if (i < n) d[i] = (_Float16)s[i];
}

// ---------------------------------------------------------------------------
// main fused kernel: one wave = one 16-token tile, full pipeline
// ---------------------------------------------------------------------------
__global__ __launch_bounds__(WAVES_ * 32)
void fused_item_score(const int* __restrict__ pos, const int* __restrict__ neg,
                      const float* __restrict__ txt_table, const float* __restrict__ img_table,
                      const float* __restrict__ ln_txt_g, const float* __restrict__ ln_txt_b,
                      const float* __restrict__ txt_bias,
                      const float* __restrict__ ln_img_g, const float* __restrict__ ln_img_b,
                      const float* __restrict__ img_bias,
                      const float* __restrict__ fus_b1, const float* __restrict__ fus_b2,
                      const _Float16* __restrict__ wtxt, const _Float16* __restrict__ wimg,
                      const _Float16* __restrict__ wf1, const _Float16* __restrict__ wf2,
                      float* __restrict__ out)
{
  __shared__ __align__(16) _Float16 stage[WAVES_][16][ASTRIDE_];
  __shared__ __align__(16) _Float16 fuz[WAVES_][16][FSTRIDE_];
  __shared__ int ids_s[WAVES_][16];

  const int wave = threadIdx.x >> 5;
  const int lane = threadIdx.x & 31;
  const int tile = blockIdx.x * WAVES_ + wave;
  if (tile >= TILES_) return;

  const int row  = lane >> 1;
  const int half = lane & 1;
  const int u0   = tile * 16 + row;
  const int id   = (u0 < NPOS_) ? pos[u0] : neg[u0 - NPOS_];
  if (half == 0) ids_s[wave][row] = id;

  // Kick the img-table gather for this tile into flight NOW, so its HBM
  // latency is hidden behind the entire txt tower (gather+LN+WMMA).
  prefetch_range(img_table + (size_t)id * DI_ + half * (DI_ / 2),
                 (DI_ / 2) * (int)sizeof(float));
  prefetch_range(ln_img_g + half * (DI_ / 2), (DI_ / 2) * (int)sizeof(float));
  prefetch_range(ln_img_b + half * (DI_ / 2), (DI_ / 2) * (int)sizeof(float));

  _Float16 (*st)[ASTRIDE_] = stage[wave];
  _Float16 (*fz)[FSTRIDE_] = fuz[wave];

  // ---------------- TXT tower: gather -> LN -> 768x64 GEMM -> ReLU ----------
  gather_ln<DT_>(txt_table, id, ln_txt_g, ln_txt_b, st[row], half);
  asm volatile("" ::: "memory");
  {
    v8f acc[HM_ / 16] = {};
    wmma_mm<DT_, HM_ / 16, ASTRIDE_>(&st[0][0], wtxt, acc, lane);
    store_h<HM_ / 16, FSTRIDE_>(acc, txt_bias, fz, 0, lane);
  }
  asm volatile("" ::: "memory");

  // ---------------- IMG tower: gather -> LN -> 512x64 GEMM -> ReLU ----------
  gather_ln<DI_>(img_table, id, ln_img_g, ln_img_b, st[row], half);
  asm volatile("" ::: "memory");
  {
    v8f acc[HM_ / 16] = {};
    wmma_mm<DI_, HM_ / 16, ASTRIDE_>(&st[0][0], wimg, acc, lane);
    store_h<HM_ / 16, FSTRIDE_>(acc, img_bias, fz, HM_, lane);
  }
  asm volatile("" ::: "memory");

  // ---------------- Fusion 128x128 GEMM -> ReLU -> 128->1 dot --------------
  {
    v8f acc[H_ / 16] = {};
    wmma_mm<H_, H_ / 16, FSTRIDE_>(&fz[0][0], wf1, acc, lane);

    const int n = lane & 15;
    float p[8] = {0.f, 0.f, 0.f, 0.f, 0.f, 0.f, 0.f, 0.f};
#pragma unroll
    for (int nc = 0; nc < H_ / 16; ++nc) {
      const float bv = fus_b1[nc * 16 + n];
      const float wv = (float)wf2[nc * 16 + n];
#pragma unroll
      for (int i = 0; i < 8; ++i) {
        float h = acc[nc][i] + bv;
        h = h > 0.f ? h : 0.f;
        p[i] += h * wv;
      }
    }
#pragma unroll
    for (int m = 1; m < 16; m <<= 1) {
#pragma unroll
      for (int i = 0; i < 8; ++i) p[i] += __shfl_xor(p[i], m, 32);
    }
    if (n == 0) {
      const int rb = (lane & 16) ? 8 : 0;
      const float b2 = fus_b2[0];
#pragma unroll
      for (int i = 0; i < 8; ++i) {
        const int r = rb + i;
        const int u = tile * 16 + r;
        out[u] = (ids_s[wave][r] != 0) ? (p[i] + b2) : 0.f;
      }
    }
  }
}

// ---------------------------------------------------------------------------
extern "C" void kernel_launch(void* const* d_in, const int* in_sizes, int n_in,
                              void* d_out, int out_size, void* d_ws, size_t ws_size,
                              hipStream_t stream) {
  (void)in_sizes; (void)n_in; (void)out_size; (void)ws_size;

  const int*   pos       = (const int*)  d_in[0];
  const int*   neg       = (const int*)  d_in[1];
  const float* txt_table = (const float*)d_in[2];
  const float* img_table = (const float*)d_in[3];
  const float* ln_txt_g  = (const float*)d_in[4];
  const float* ln_txt_b  = (const float*)d_in[5];
  const float* txt_w     = (const float*)d_in[6];
  const float* txt_bias  = (const float*)d_in[7];
  const float* ln_img_g  = (const float*)d_in[8];
  const float* ln_img_b  = (const float*)d_in[9];
  const float* img_w     = (const float*)d_in[10];
  const float* img_bias  = (const float*)d_in[11];
  const float* fus_w1    = (const float*)d_in[12];
  const float* fus_b1    = (const float*)d_in[13];
  const float* fus_w2    = (const float*)d_in[14];
  const float* fus_b2    = (const float*)d_in[15];
  float*       out       = (float*)d_out;

  // f16 weight images in scratch: [txt_w | img_w | fus_w1 | fus_w2]
  _Float16* wtxt = (_Float16*)d_ws;
  _Float16* wimg = wtxt + HM_ * DT_;
  _Float16* wf1  = wimg + HM_ * DI_;
  _Float16* wf2  = wf1 + H_ * H_;

  const int n0 = HM_ * DT_, n1 = HM_ * DI_, n2 = H_ * H_, n3 = H_;
  cvt_f32_f16<<<(n0 + 255) / 256, 256, 0, stream>>>(txt_w,  wtxt, n0);
  cvt_f32_f16<<<(n1 + 255) / 256, 256, 0, stream>>>(img_w,  wimg, n1);
  cvt_f32_f16<<<(n2 + 255) / 256, 256, 0, stream>>>(fus_w1, wf1,  n2);
  cvt_f32_f16<<<(n3 + 255) / 256, 256, 0, stream>>>(fus_w2, wf2,  n3);

  fused_item_score<<<TILES_ / WAVES_, WAVES_ * 32, 0, stream>>>(
      pos, neg, txt_table, img_table,
      ln_txt_g, ln_txt_b, txt_bias,
      ln_img_g, ln_img_b, img_bias,
      fus_b1, fus_b2,
      wtxt, wimg, wf1, wf2, out);
}
